// SAGE_50577534878115
// MI455X (gfx1250) — matrix operand order repaired and checked by reading, not compiled, not run on hardware
//
#include <hip/hip_runtime.h>

typedef __attribute__((ext_vector_type(16))) __bf16 v16bf;
typedef __attribute__((ext_vector_type(8)))  float  v8f;
typedef __attribute__((ext_vector_type(4)))  float  v4f;

#define DIM_IN 128   // layer inputs are always 128 wide (D_IN = D_H = 128)

// ---------------------------------------------------------------------------
// Utility kernels
// ---------------------------------------------------------------------------
__global__ void zero_f32(float* __restrict__ p, long long n) {
  long long i = (long long)blockIdx.x * blockDim.x + threadIdx.x;
  long long stride = (long long)gridDim.x * blockDim.x;
  for (; i < n; i += stride) p[i] = 0.0f;
}

// Pack fp32 weights [K=128][ncols] into bf16 B-fragment order:
//   Wp[(((ct*4 + t)*32 + lane)*16 + j)] = bf16( W[k][n] )
//   with k = t*32 + (lane>>4)*8 + (j&7) + (j>>3)*16,  n = ct*16 + (lane&15)
// so each lane's 16-value fragment is 32 contiguous bytes (one b128 pair).
__global__ void pack_b(const float* __restrict__ W, __bf16* __restrict__ Wp, int ncols) {
  const int total = ncols * 128;
  int i = blockIdx.x * blockDim.x + threadIdx.x;
  if (i >= total) return;
  const int j     = i & 15;
  const int lane  = (i >> 4) & 31;
  const int blkid = i >> 9;        // ct*4 + t
  const int t     = blkid & 3;
  const int ct    = blkid >> 2;
  const int k = t * 32 + (lane >> 4) * 8 + (j & 7) + (j >> 3) * 16;
  const int n = ct * 16 + (lane & 15);
  Wp[i] = (__bf16)W[(size_t)k * ncols + n];
}

// in-degree histogram (float, so the mean divide is a cheap reciprocal later)
__global__ void degree_k(const long long* __restrict__ dst, float* __restrict__ deg, int E) {
  int e = blockIdx.x * blockDim.x + threadIdx.x;
  if (e < E) atomicAdd(&deg[(int)dst[e]], 1.0f);
}

// one wave32 per edge: lane l moves feature chunk [4l, 4l+4) -> L2-resident atomics
__global__ void scatter_k(const float* __restrict__ feat, const long long* __restrict__ src,
                          const long long* __restrict__ dst, float* __restrict__ agg, int E) {
  const int lane = threadIdx.x & 31;
  const int e = blockIdx.x * (blockDim.x >> 5) + (threadIdx.x >> 5);
  if (e >= E) return;
  const int s = (int)src[e];
  const int d = (int)dst[e];
  const float4 v = *(const float4*)(feat + (size_t)s * DIM_IN + lane * 4);
  float* o = agg + (size_t)d * DIM_IN + lane * 4;
  atomicAdd(o + 0, v.x);
  atomicAdd(o + 1, v.y);
  atomicAdd(o + 2, v.z);
  atomicAdd(o + 3, v.w);
}

// ---------------------------------------------------------------------------
// A fragment (16x32 bf16, ISA 7.12.2): lane holds row m; two 8-element runs
// at k-offsets [ka, ka+8) and [ka+16, ka+24). Explicit b128 loads + cvt.
// ---------------------------------------------------------------------------
__device__ __forceinline__ v16bf pack_a(const float* __restrict__ p, float s) {
  const v4f f0 = *(const v4f*)(p);
  const v4f f1 = *(const v4f*)(p + 4);
  const v4f f2 = *(const v4f*)(p + 16);
  const v4f f3 = *(const v4f*)(p + 20);
  v16bf r;
  r[0]  = (__bf16)(f0.x * s); r[1]  = (__bf16)(f0.y * s);
  r[2]  = (__bf16)(f0.z * s); r[3]  = (__bf16)(f0.w * s);
  r[4]  = (__bf16)(f1.x * s); r[5]  = (__bf16)(f1.y * s);
  r[6]  = (__bf16)(f1.z * s); r[7]  = (__bf16)(f1.w * s);
  r[8]  = (__bf16)(f2.x * s); r[9]  = (__bf16)(f2.y * s);
  r[10] = (__bf16)(f2.z * s); r[11] = (__bf16)(f2.w * s);
  r[12] = (__bf16)(f3.x * s); r[13] = (__bf16)(f3.y * s);
  r[14] = (__bf16)(f3.z * s); r[15] = (__bf16)(f3.w * s);
  return r;
}

// ---------------------------------------------------------------------------
// Fused SAGE layer GEMM:
//   Out[N, NCOLS] = act( (Agg * (1/max(deg,1))) @ Wl  +  Self @ Wr )
// One wave computes a 16-row strip. A fragments (4 k-chunks x 2 operands)
// are built once per wave; the column-tile loop is pure B-load + WMMA.
// Epilogue: wave-uniform tail test so the common path is unpredicated stores.
// ---------------------------------------------------------------------------
template <int NCOLS, bool RELU>
__global__ void sage_gemm(const float* __restrict__ Agg, const float* __restrict__ deg,
                          const float* __restrict__ Self, const __bf16* __restrict__ WlP,
                          const __bf16* __restrict__ WrP, float* __restrict__ Out, int N) {
  const int lane = threadIdx.x & 31;
  const int wavesPerBlk = blockDim.x >> 5;
  const int rowBase = (blockIdx.x * wavesPerBlk + (threadIdx.x >> 5)) * 16;
  if (rowBase >= N) return;  // wave-uniform exit: EXEC stays all-1s for WMMA

  const int m  = lane & 15;
  const int hi = lane >> 4;

  int row = rowBase + m;
  if (row >= N) row = N - 1;  // clamp loads; stores handle the tail
  const float* aggRow  = Agg  + (size_t)row * DIM_IN;
  const float* selfRow = Self + (size_t)row * DIM_IN;
  const float  invdeg  = 1.0f / fmaxf(deg[row], 1.0f);

  // Hoisted A fragments: independent of the output column tile.
  v16bf aL[4], aR[4];
#pragma unroll
  for (int t = 0; t < 4; ++t) {
    const int ka = t * 32 + hi * 8;
    aL[t] = pack_a(aggRow + ka, invdeg);
    aR[t] = pack_a(selfRow + ka, 1.0f);
  }

  const bool fullTile = (rowBase + 16 <= N);  // wave-uniform
  const int  r0 = rowBase + hi * 8;           // C/D: VGPR v -> row r0 + v
  float* outBase = Out + (size_t)r0 * NCOLS + m;

#pragma unroll
  for (int ct = 0; ct < NCOLS / 16; ++ct) {
    v8f acc = {0.f, 0.f, 0.f, 0.f, 0.f, 0.f, 0.f, 0.f};
#pragma unroll
    for (int t = 0; t < 4; ++t) {
      // packed fragment: 16 contiguous bf16 per lane (32B, coalesced)
      const size_t fo = (((size_t)(ct * 4 + t)) * 32 + lane) * 16;
      const v16bf bL = *(const v16bf*)(WlP + fo);
      acc = __builtin_amdgcn_wmma_f32_16x16x32_bf16(false, aL[t], false, bL,
                                                    (short)0, acc, false, false);
      const v16bf bR = *(const v16bf*)(WrP + fo);
      acc = __builtin_amdgcn_wmma_f32_16x16x32_bf16(false, aR[t], false, bR,
                                                    (short)0, acc, false, false);
    }
    float* outp = outBase + ct * 16;
    if (fullTile) {
      // common path: 8 plain stores, compile-time row offsets, no exec churn
#pragma unroll
      for (int v = 0; v < 8; ++v) {
        float val = acc[v];
        if (RELU) val = fmaxf(val, 0.0f);
        outp[(size_t)v * NCOLS] = val;
      }
    } else {
      // tail wave only: per-row predicated stores
#pragma unroll
      for (int v = 0; v < 8; ++v) {
        float val = acc[v];
        if (RELU) val = fmaxf(val, 0.0f);
        if (r0 + v < N) outp[(size_t)v * NCOLS] = val;
      }
    }
  }
}

// ---------------------------------------------------------------------------
// Host-side orchestration
// ---------------------------------------------------------------------------
extern "C" void kernel_launch(void* const* d_in, const int* in_sizes, int n_in,
                              void* d_out, int out_size, void* d_ws, size_t ws_size,
                              hipStream_t stream) {
  const float* x = (const float*)d_in[0];
  const long long* ei = (const long long*)d_in[1];  // int64 per reference
  const float* W1_l = (const float*)d_in[2];
  const float* W1_r = (const float*)d_in[3];
  const float* W2_l = (const float*)d_in[4];
  const float* W2_r = (const float*)d_in[5];
  float* out = (float*)d_out;

  const int N = in_sizes[0] / DIM_IN;
  const int E = in_sizes[1] / 2;
  const long long* src = ei;
  const long long* dst = ei + E;

  // workspace layout (256B-aligned float regions, packed bf16 weights after)
  char* ws = (char*)d_ws;
  float* deg = (float*)ws;
  size_t off = ((size_t)N * 4 + 255) & ~(size_t)255;
  float* agg = (float*)(ws + off);                 // N*128, reused by both layers
  float* h   = agg + (size_t)N * DIM_IN;           // N*128
  __bf16* w1l = (__bf16*)(h + (size_t)N * DIM_IN); // 256B-aligned (N*512 % 256 == 0)
  __bf16* w1r = w1l + 128 * 128;
  __bf16* w2l = w1r + 128 * 128;
  __bf16* w2r = w2l + 128 * 64;

  const int ZB = 4096;
  // zero degree + accumulator
  zero_f32<<<ZB, 256, 0, stream>>>(deg, (long long)N);
  zero_f32<<<ZB, 256, 0, stream>>>(agg, (long long)N * DIM_IN);
  // bf16 weight packing into B-fragment order (tiny, L2-resident)
  pack_b<<<(128 * 128 + 255) / 256, 256, 0, stream>>>(W1_l, w1l, 128);
  pack_b<<<(128 * 128 + 255) / 256, 256, 0, stream>>>(W1_r, w1r, 128);
  pack_b<<<(128 * 64 + 255) / 256, 256, 0, stream>>>(W2_l, w2l, 64);
  pack_b<<<(128 * 64 + 255) / 256, 256, 0, stream>>>(W2_r, w2r, 64);
  // in-degrees
  degree_k<<<(E + 255) / 256, 256, 0, stream>>>(dst, deg, E);

  // layer 1: scatter-sum x, then h = relu(mean @ W1_l + x @ W1_r)
  scatter_k<<<(E + 7) / 8, 256, 0, stream>>>(x, src, dst, agg, E);
  sage_gemm<128, true><<<(N + 127) / 128, 256, 0, stream>>>(agg, deg, x, w1l, w1r, h, N);

  // layer 2: re-zero accumulator, scatter-sum h, out = mean @ W2_l + h @ W2_r
  zero_f32<<<ZB, 256, 0, stream>>>(agg, (long long)N * DIM_IN);
  scatter_k<<<(E + 7) / 8, 256, 0, stream>>>(h, src, dst, agg, E);
  sage_gemm<64, false><<<(N + 127) / 128, 256, 0, stream>>>(agg, deg, h, w2l, w2r, out, N);
}
